// Net_76347338654180
// MI455X (gfx1250) — 8-bit matrix operand order repaired and checked by reading, compiled, not run on hardware
//
#include <hip/hip_runtime.h>
#include <math.h>


#define F_IN  512
#define HID1  128
#define NCLS  32

typedef __attribute__((ext_vector_type(8))) int v8i;

__device__ __forceinline__ unsigned fenc(float f) {
  unsigned u = __float_as_uint(f);
  return (u & 0x80000000u) ? ~u : (u | 0x80000000u);
}
__device__ __forceinline__ float fdec(unsigned u) {
  u = (u & 0x80000000u) ? (u & 0x7FFFFFFFu) : ~u;
  return __uint_as_float(u);
}
__device__ __forceinline__ signed char sgn8(float v) {
  return (v > 0.f) ? (signed char)1 : ((v < 0.f) ? (signed char)-1 : (signed char)0);
}

__global__ void absmean_k(const float* __restrict__ W1, int n1,
                          const float* __restrict__ W2, int n2,
                          float* __restrict__ scale) {
  __shared__ float red[256];
  const float* W = (blockIdx.x == 0) ? W1 : W2;
  int n = (blockIdx.x == 0) ? n1 : n2;
  float s = 0.f;
  for (int i = threadIdx.x; i < n; i += blockDim.x) s += fabsf(W[i]);
  red[threadIdx.x] = s;
  __syncthreads();
  for (int st = 128; st > 0; st >>= 1) {
    if ((int)threadIdx.x < st) red[threadIdx.x] += red[threadIdx.x + st];
    __syncthreads();
  }
  if (threadIdx.x == 0) scale[blockIdx.x] = red[0] / (float)n;
}

__global__ void pack_b_k(const float* __restrict__ W, int* __restrict__ Bpack,
                         int KDIM, int NDIM) {
  int t = blockIdx.x * blockDim.x + threadIdx.x;
  int total = (KDIM >> 6) * (NDIM >> 4) * 32;
  if (t >= total) return;
  int lane = t & 31;
  int grp  = t >> 5;
  int ntiles = NDIM >> 4;
  int nt = grp % ntiles;
  int ks = grp / ntiles;
  int n  = nt * 16 + (lane & 15);
  int kb = ks * 64 + ((lane >> 4) * 8);
#pragma unroll
  for (int j = 0; j < 8; ++j) {
    int kk = kb + ((j >> 1) * 16) + ((j & 1) * 4);
    unsigned dv = 0;
    for (int b = 0; b < 4; ++b) {
      float w = W[(size_t)(kk + b) * NDIM + n];
      dv |= ((unsigned)(unsigned char)sgn8(w)) << (8 * b);
    }
    Bpack[(size_t)t * 8 + j] = (int)dv;
  }
}

__global__ void bn_sign_pack_k(const float* __restrict__ x,
                               const float* __restrict__ mean,
                               const float* __restrict__ var,
                               signed char* __restrict__ out, int total) {
  int i = blockIdx.x * blockDim.x + threadIdx.x;
  if (i * 4 >= total) return;
  int f0 = (i * 4) & (F_IN - 1);
  const float* xp = x + (size_t)i * 4;
  unsigned dv = 0;
#pragma unroll
  for (int b = 0; b < 4; ++b) {
    float v = (xp[b] - mean[f0 + b]) * rsqrtf(var[f0 + b] + 1e-5f);
    dv |= ((unsigned)(unsigned char)sgn8(v)) << (8 * b);
  }
  ((unsigned*)out)[i] = dv;
}

__global__ void sign_pack_k(const float* __restrict__ H,
                            const float* __restrict__ bias,
                            signed char* __restrict__ out, int total, int D) {
  int i = blockIdx.x * blockDim.x + threadIdx.x;
  if (i * 4 >= total) return;
  int f0 = (i * 4) % D;
  unsigned dv = 0;
#pragma unroll
  for (int b = 0; b < 4; ++b) {
    float v = H[(size_t)i * 4 + b] + bias[f0 + b];
    dv |= ((unsigned)(unsigned char)sgn8(v)) << (8 * b);
  }
  ((unsigned*)out)[i] = dv;
}

template <int KSTEPS, int NTILES, int KDIM, int NDIM>
__global__ __launch_bounds__(NTILES * 32) void bin_gemm_wmma_k(
    const signed char* __restrict__ Abin, const int* __restrict__ Bpack,
    const float* __restrict__ scale, float* __restrict__ H) {
  const int wave = threadIdx.x >> 5;
  const int lane = threadIdx.x & 31;
  const int mtile = blockIdx.x;
  const int ko = (lane >> 4) * 8;
  const signed char* arow = Abin + (size_t)(mtile * 16 + (lane & 15)) * KDIM;
  v8i acc = {0, 0, 0, 0, 0, 0, 0, 0};
  for (int ks = 0; ks < KSTEPS; ++ks) {
    const int* ap = (const int*)(arow + ks * 64 + ko);
    v8i a;
    a[0] = ap[0];  a[1] = ap[1];
    a[2] = ap[4];  a[3] = ap[5];
    a[4] = ap[8];  a[5] = ap[9];
    a[6] = ap[12]; a[7] = ap[13];
    const v8i* bp =
        (const v8i*)(Bpack + ((size_t)((ks * NTILES + wave) * 32 + lane)) * 8);
    v8i b = *bp;
    acc = __builtin_amdgcn_wmma_i32_16x16x64_iu8(true, a, true, b, acc,
                                                 false, false);
  }
  const float sc = scale[0];
  const int n = lane & 15;
  const int mbase = (lane >> 4) * 8;
  float* out = H + (size_t)(mtile * 16 + mbase) * NDIM + wave * 16 + n;
#pragma unroll
  for (int j = 0; j < 8; ++j) out[(size_t)j * NDIM] = sc * (float)acc[j];
}

__global__ void attn_dots_k(const float* __restrict__ H,
                            const float* __restrict__ asrc,
                            const float* __restrict__ adst,
                            float* __restrict__ al, float* __restrict__ ar,
                            int Nn, int heads, int C) {
  int t = blockIdx.x * blockDim.x + threadIdx.x;
  if (t >= Nn * heads) return;
  int h = t % heads;
  const float* hp = H + (size_t)(t / heads) * heads * C + (size_t)h * C;
  float sa = 0.f, sd = 0.f;
  for (int c = 0; c < C; ++c) {
    float v = hp[c];
    sa += v * asrc[h * C + c];
    sd += v * adst[h * C + c];
  }
  al[t] = sa;
  ar[t] = sd;
}

__global__ void fill_u32_k(unsigned* __restrict__ p, unsigned v, int n) {
  int i = blockIdx.x * blockDim.x + threadIdx.x;
  if (i < n) p[i] = v;
}

__global__ void edge_max_k(const int* __restrict__ ei, int E, int ET,
                           const float* __restrict__ al,
                           const float* __restrict__ ar,
                           unsigned* __restrict__ m, int heads) {
  int t = blockIdx.x * blockDim.x + threadIdx.x;
  if (t >= ET * heads) return;
  int e = t / heads, h = t - e * heads;
  int s, d;
  if (e < E) { s = ei[e]; d = ei[E + e]; } else { s = e - E; d = s; }
  float v = al[s * heads + h] + ar[d * heads + h];
  v = (v > 0.f) ? v : 0.2f * v;
  atomicMax(&m[d * heads + h], fenc(v));
}

__global__ void edge_sum_k(const int* __restrict__ ei, int E, int ET,
                           const float* __restrict__ al,
                           const float* __restrict__ ar,
                           const unsigned* __restrict__ m,
                           float* __restrict__ ssum, int heads) {
  int t = blockIdx.x * blockDim.x + threadIdx.x;
  if (t >= ET * heads) return;
  int e = t / heads, h = t - e * heads;
  int s, d;
  if (e < E) { s = ei[e]; d = ei[E + e]; } else { s = e - E; d = s; }
  float v = al[s * heads + h] + ar[d * heads + h];
  v = (v > 0.f) ? v : 0.2f * v;
  atomicAdd(&ssum[d * heads + h], __expf(v - fdec(m[d * heads + h])));
}

__global__ void edge_agg_k(const int* __restrict__ ei, int E, int ET,
                           const float* __restrict__ al,
                           const float* __restrict__ ar,
                           const unsigned* __restrict__ m,
                           const float* __restrict__ ssum,
                           const float* __restrict__ H,
                           float* __restrict__ out, int heads, int C) {
  int t = blockIdx.x * blockDim.x + threadIdx.x;
  if (t >= ET * heads) return;
  int e = t / heads, h = t - e * heads;
  int s, d;
  if (e < E) { s = ei[e]; d = ei[E + e]; } else { s = e - E; d = s; }
  float v = al[s * heads + h] + ar[d * heads + h];
  v = (v > 0.f) ? v : 0.2f * v;
  float w = __expf(v - fdec(m[d * heads + h])) / ssum[d * heads + h];
  const float* hs = H + (size_t)(s * heads + h) * C;
  float* od = out + (size_t)(d * heads + h) * C;
  for (int c = 0; c < C; ++c) atomicAdd(&od[c], w * hs[c]);
}

__global__ void logsoftmax32_k(float* __restrict__ out,
                               const float* __restrict__ bias, int Nn) {
  int node = blockIdx.x * (blockDim.x >> 5) + (threadIdx.x >> 5);
  int c = threadIdx.x & 31;
  if (node >= Nn) return;
  float v = out[(size_t)node * 32 + c] + bias[c];
  float mx = v;
#pragma unroll
  for (int o = 16; o > 0; o >>= 1) mx = fmaxf(mx, __shfl_xor(mx, o, 32));
  float ex = __expf(v - mx);
  float sm = ex;
#pragma unroll
  for (int o = 16; o > 0; o >>= 1) sm += __shfl_xor(sm, o, 32);
  out[(size_t)node * 32 + c] = v - mx - __logf(sm);
}

extern "C" void kernel_launch(void* const* d_in, const int* in_sizes, int n_in,
                              void* d_out, int out_size, void* d_ws,
                              size_t ws_size, hipStream_t stream) {
  (void)n_in; (void)out_size; (void)ws_size;
  const float* x     = (const float*)d_in[0];
  const int*   ei    = (const int*)d_in[1];
  const float* rmean = (const float*)d_in[2];
  const float* rvar  = (const float*)d_in[3];
  const float* W1    = (const float*)d_in[4];
  const float* a1s   = (const float*)d_in[5];
  const float* a1d   = (const float*)d_in[6];
  const float* b1    = (const float*)d_in[7];
  const float* W2    = (const float*)d_in[8];
  const float* a2s   = (const float*)d_in[9];
  const float* a2d   = (const float*)d_in[10];
  const float* b2    = (const float*)d_in[11];

  const int Nn = in_sizes[0] / F_IN;
  const int Ee = in_sizes[1] / 2;
  const int ET = Ee + Nn;
  const int MT = Nn / 16;

  char* ws = (char*)d_ws;
  size_t off = 0;
  auto carve = [&](size_t bytes) -> void* {
    void* p = ws + off;
    off = (off + bytes + 255) & ~(size_t)255;
    return p;
  };
  float*       scale  = (float*)carve(2 * sizeof(float));
  int*         W1pack = (int*)carve((size_t)8 * 8 * 32 * 8 * sizeof(int));
  int*         W2pack = (int*)carve((size_t)2 * 2 * 32 * 8 * sizeof(int));
  signed char* hbin1  = (signed char*)carve((size_t)Nn * F_IN);
  float*       h1     = (float*)carve((size_t)Nn * HID1 * sizeof(float));
  float*       al1    = (float*)carve((size_t)Nn * 2 * sizeof(float));
  float*       ar1    = (float*)carve((size_t)Nn * 2 * sizeof(float));
  unsigned*    m1     = (unsigned*)carve((size_t)Nn * 2 * sizeof(unsigned));
  float*       s1     = (float*)carve((size_t)Nn * 2 * sizeof(float));
  signed char* hbin2  = (signed char*)carve((size_t)Nn * HID1);
  float*       h2     = (float*)carve((size_t)Nn * NCLS * sizeof(float));
  float*       al2    = (float*)carve((size_t)Nn * sizeof(float));
  float*       ar2    = (float*)carve((size_t)Nn * sizeof(float));
  unsigned*    m2     = (unsigned*)carve((size_t)Nn * sizeof(unsigned));
  float*       s2     = (float*)carve((size_t)Nn * sizeof(float));
  float*       out1   = (float*)hbin1;
  float*       out2   = (float*)d_out;

  const int TB = 256;
  auto grid = [&](long long work) { return (int)((work + TB - 1) / TB); };

  absmean_k<<<2, TB, 0, stream>>>(W1, F_IN * HID1, W2, HID1 * NCLS, scale);
  pack_b_k<<<grid(8 * 8 * 32), TB, 0, stream>>>(W1, W1pack, F_IN, HID1);
  pack_b_k<<<grid(2 * 2 * 32), TB, 0, stream>>>(W2, W2pack, HID1, NCLS);

  bn_sign_pack_k<<<grid((long long)Nn * F_IN / 4), TB, 0, stream>>>(
      x, rmean, rvar, hbin1, Nn * F_IN);
  bin_gemm_wmma_k<8, 8, F_IN, HID1><<<MT, 8 * 32, 0, stream>>>(
      hbin1, W1pack, scale, h1);
  attn_dots_k<<<grid((long long)Nn * 2), TB, 0, stream>>>(
      h1, a1s, a1d, al1, ar1, Nn, 2, 64);
  fill_u32_k<<<grid((long long)Nn * 2), TB, 0, stream>>>(m1, 0u, Nn * 2);
  fill_u32_k<<<grid((long long)Nn * 2), TB, 0, stream>>>((unsigned*)s1, 0u, Nn * 2);
  fill_u32_k<<<grid((long long)Nn * HID1), TB, 0, stream>>>((unsigned*)out1, 0u,
                                                            Nn * HID1);
  edge_max_k<<<grid((long long)ET * 2), TB, 0, stream>>>(ei, Ee, ET, al1, ar1, m1, 2);
  edge_sum_k<<<grid((long long)ET * 2), TB, 0, stream>>>(ei, Ee, ET, al1, ar1, m1, s1, 2);
  edge_agg_k<<<grid((long long)ET * 2), TB, 0, stream>>>(ei, Ee, ET, al1, ar1, m1, s1,
                                                         h1, out1, 2, 64);

  sign_pack_k<<<grid((long long)Nn * HID1 / 4), TB, 0, stream>>>(
      out1, b1, hbin2, Nn * HID1, HID1);
  bin_gemm_wmma_k<2, 2, HID1, NCLS><<<MT, 2 * 32, 0, stream>>>(
      hbin2, W2pack, scale + 1, h2);
  attn_dots_k<<<grid((long long)Nn), TB, 0, stream>>>(
      h2, a2s, a2d, al2, ar2, Nn, 1, NCLS);
  fill_u32_k<<<grid((long long)Nn), TB, 0, stream>>>(m2, 0u, Nn);
  fill_u32_k<<<grid((long long)Nn), TB, 0, stream>>>((unsigned*)s2, 0u, Nn);
  fill_u32_k<<<grid((long long)Nn * NCLS), TB, 0, stream>>>((unsigned*)out2, 0u,
                                                            Nn * NCLS);
  edge_max_k<<<grid((long long)ET), TB, 0, stream>>>(ei, Ee, ET, al2, ar2, m2, 1);
  edge_sum_k<<<grid((long long)ET), TB, 0, stream>>>(ei, Ee, ET, al2, ar2, m2, s2, 1);
  edge_agg_k<<<grid((long long)ET), TB, 0, stream>>>(ei, Ee, ET, al2, ar2, m2, s2,
                                                     h2, out2, 1, NCLS);

  logsoftmax32_k<<<(Nn + 7) / 8, TB, 0, stream>>>(out2, b2, Nn);
}
